// QuantumConvolution_43671227466089
// MI455X (gfx1250) — matrix operand order repaired and checked by reading, not compiled, not run on hardware
//
#include <hip/hip_runtime.h>
#include <hip/hip_bf16.h>
#include <math.h>

// Problem constants (match setup_inputs)
// B=32, Cin=16, H=W=64, Cout=8, L=3, fs=4, st=2, oh=ow=31, D=dim=256, n=8
#define NPATCH 30752          // 32*31*31
#define NTILES 1922           // NPATCH/16
#define WPC    32             // workgroups per channel in main kernel

typedef __attribute__((ext_vector_type(16))) _Float16 v16h;
typedef __attribute__((ext_vector_type(8)))  float    v8f;

union HF16 {
    v16h v;
    _Float16 h[16];
    uint4 q[2];
};

// ---------------------------------------------------------------------------
// Kernel 1: build per-channel circuit unitary U_c (256x256 complex, fp32)
// by evolving 32 basis columns per block through the circuit in LDS.
// grid = 64 blocks (8 channels x 8 column-chunks), 256 threads.
// Ur/Ui layout: [c][k][j]  (k = output amplitude index, j = input basis)
// ---------------------------------------------------------------------------
__global__ __launch_bounds__(256) void k_build_unitary(
        const float* __restrict__ w,   // (8,3,8,3)
        float* __restrict__ Ur, float* __restrict__ Ui) {
    __shared__ float sre[32 * 256];
    __shared__ float sim[32 * 256];
    const int blk   = blockIdx.x;
    const int ch    = blk >> 3;
    const int chunk = blk & 7;
    const int t     = threadIdx.x;   // 0..255

    // init: column (chunk*32+col) = basis vector e_j
    for (int col = 0; col < 32; ++col) {
        int j = chunk * 32 + col;
        sre[col * 256 + t] = (t == j) ? 1.f : 0.f;
        sim[col * 256 + t] = 0.f;
    }
    __syncthreads();

    const float* wc = w + ch * 3 * 8 * 3;
    for (int l = 0; l < 3; ++l) {
        // single-qubit rotations
        for (int q = 0; q < 8; ++q) {
            float phi = wc[(l * 8 + q) * 3 + 0];
            float th  = wc[(l * 8 + q) * 3 + 1];
            float om  = wc[(l * 8 + q) * 3 + 2];
            float c  = cosf(0.5f * th), s = sinf(0.5f * th);
            float a  = 0.5f * (phi + om), bb = 0.5f * (phi - om);
            float ca = cosf(a), sa = sinf(a);
            float cb = cosf(bb), sb = sinf(bb);
            float m00r =  ca * c, m00i = -sa * c;
            float m01r = -cb * s, m01i = -sb * s;
            float m10r =  cb * s, m10i = -sb * s;
            float m11r =  ca * c, m11i =  sa * c;
            const int bitpos = 7 - q;
            const int stride = 1 << bitpos;
            // 32 cols * 128 pairs = 4096 pairs; 16 per thread
            for (int r = 0; r < 16; ++r) {
                int pp  = t + 256 * r;
                int col = pp >> 7;
                int pr  = pp & 127;
                int i0  = ((pr >> bitpos) << (bitpos + 1)) | (pr & (stride - 1));
                int i1  = i0 + stride;
                int base = col * 256;
                float x0r = sre[base + i0], x0i = sim[base + i0];
                float x1r = sre[base + i1], x1i = sim[base + i1];
                sre[base + i0] = m00r * x0r - m00i * x0i + m01r * x1r - m01i * x1i;
                sim[base + i0] = m00r * x0i + m00i * x0r + m01r * x1i + m01i * x1r;
                sre[base + i1] = m10r * x0r - m10i * x0i + m11r * x1r - m11i * x1i;
                sim[base + i1] = m10r * x0i + m10i * x0r + m11r * x1i + m11i * x1r;
            }
            __syncthreads();
        }
        // CNOT ring: ctrl=q (bit 7-q), target=(q+1)%8 (bit 7-((q+1)%8));
        // perm = swap of {idx, idx^tmask} where ctrl bit is set.
        for (int q = 0; q < 8; ++q) {
            int cbit = 7 - q;
            int tbit = 7 - ((q + 1) & 7);
            int lo = cbit < tbit ? cbit : tbit;
            int hi = cbit < tbit ? tbit : cbit;
            int cmask = 1 << cbit, tmask = 1 << tbit;
            // 32 cols * 64 swaps = 2048; 8 per thread
            for (int r = 0; r < 8; ++r) {
                int ss  = t + 256 * r;
                int col = ss >> 6;
                int u   = ss & 63;
                int low  = u & ((1 << lo) - 1);
                int mid  = (u >> lo) & ((1 << (hi - 1 - lo)) - 1);
                int high = u >> (hi - 1);
                int idx  = low | (mid << (lo + 1)) | (high << (hi + 1));
                idx |= cmask;                 // ctrl=1, target=0
                int idx2 = idx ^ tmask;
                int base = col * 256;
                float tr = sre[base + idx],  ti = sim[base + idx];
                sre[base + idx]  = sre[base + idx2];
                sim[base + idx]  = sim[base + idx2];
                sre[base + idx2] = tr;
                sim[base + idx2] = ti;
            }
            __syncthreads();
        }
    }
    // write U[c][k=t][j]
    for (int col = 0; col < 32; ++col) {
        int j = chunk * 32 + col;
        Ur[(ch * 256 + t) * 256 + j] = sre[col * 256 + t];
        Ui[(ch * 256 + t) * 256 + j] = sim[col * 256 + t];
    }
}

// ---------------------------------------------------------------------------
// Kernel 2: A_c[i][j] = sum_k sign_k (Ur[k,i]Ur[k,j] + Ui[k,i]Ui[k,j]), f16.
// sign_k = +1 for k<128, -1 otherwise (Z on qubit 0). A is symmetric.
// grid = 8*256 blocks (channel,i), 256 threads (j).
// ---------------------------------------------------------------------------
__global__ __launch_bounds__(256) void k_build_A(
        const float* __restrict__ Ur, const float* __restrict__ Ui,
        _Float16* __restrict__ A) {
    const int blk = blockIdx.x;
    const int ch  = blk >> 8;
    const int i   = blk & 255;
    const int j   = threadIdx.x;
    const float* ur = Ur + ch * 65536;
    const float* ui = Ui + ch * 65536;
    float acc = 0.f;
    #pragma unroll 4
    for (int k = 0; k < 256; ++k) {
        float sgn = (k < 128) ? 1.f : -1.f;
        acc += sgn * (ur[k * 256 + i] * ur[k * 256 + j] +
                      ui[k * 256 + i] * ui[k * 256 + j]);
    }
    A[(ch * 256 + i) * 256 + j] = (_Float16)acc;
}

// ---------------------------------------------------------------------------
// Kernel 3: out[b,c,oy,ox] = (p^T A_c p)/||p||^2 + bias[c], WMMA f16 GEMM.
// grid = 8*WPC workgroups x 256 threads (8 waves). Each WG owns one channel,
// stages A_c (128KB f16) in LDS once; each wave processes 16-patch tiles.
// Dynamic LDS: A(131072) + 8 waves * P-tile(8192) + 8 * norms(64) = 197120 B.
// ---------------------------------------------------------------------------
__global__ __launch_bounds__(256) void k_qconv(
        const float* __restrict__ x,        // (32,16,64,64)
        const _Float16* __restrict__ A,     // (8,256,256)
        const float* __restrict__ bias,     // (8)
        float* __restrict__ out) {          // (32,8,31,31)
    extern __shared__ char lds[];
    _Float16* Al = (_Float16*)lds;                                  // 256x256
    const int tid  = threadIdx.x;
    const int wave = tid >> 5;
    const int lane = tid & 31;
    const int half = lane >> 4;      // 0: lanes 0-15, 1: lanes 16-31
    const int row  = lane & 15;
    const int wg   = blockIdx.x;
    const int ch   = wg / WPC;
    const int wgl  = wg % WPC;
    _Float16* Pl  = (_Float16*)(lds + 131072 + wave * 8192);        // 16x256 f16
    float*    nrm = (float*)(lds + 131072 + 65536 + wave * 64);     // 16 f32

    // cooperatively stage A_c into LDS (8192 x uint4 = 128KB)
    {
        const uint4* src = (const uint4*)(A + ch * 65536);
        uint4* dst = (uint4*)Al;
        for (int it = tid; it < 8192; it += 256) dst[it] = src[it];
    }
    __syncthreads();
    const float bch = bias[ch];

    for (int tile = wgl * 8 + wave; tile < NTILES; tile += WPC * 8) {
        // ---- stage 16 patches (f16) + fp32 norms; 2 lanes per patch row ----
        {
            int prow = lane >> 1;
            int p = tile * 16 + prow;
            int b = p / 961;
            int rem = p - b * 961;
            int oy = rem / 31;
            int ox = rem - oy * 31;
            const float* xb = x + (size_t)b * 65536 + (size_t)(oy * 2) * 64 + (ox * 2);
            float n2 = 0.f;
            int jb = (lane & 1) * 128;
            #pragma unroll 8
            for (int tj = 0; tj < 128; ++tj) {
                int j = jb + tj;
                int cin = j >> 4, fy = (j >> 2) & 3, fx = j & 3;
                float v = xb[(size_t)cin * 4096 + fy * 64 + fx];
                n2 += v * v;
                Pl[prow * 256 + j] = (_Float16)v;
            }
            n2 += __shfl_xor(n2, 1, 32);
            if ((lane & 1) == 0) nrm[prow] = n2;
        }

        // ---- load 8 K-step A-operand fragments (patch tile, 16x32 each) ----
        // ISA layout: lane<16: elts 0-7 -> K 0-7, elts 8-15 -> K 16-23;
        //             lane>=16: +8. Row M = lane%16.
        v16h Pf[8];
        #pragma unroll
        for (int ks = 0; ks < 8; ++ks) {
            HF16 a;
            #pragma unroll
            for (int e = 0; e < 16; ++e) {
                int k = ks * 32 + (e < 8 ? e : e + 8) + (half ? 8 : 0);
                a.h[e] = Pl[row * 256 + k];
            }
            Pf[ks] = a.v;
        }

        float part[8] = {0.f, 0.f, 0.f, 0.f, 0.f, 0.f, 0.f, 0.f};

        // ---- 16 i-tiles: T(16x16) = P(16x256) x A(256x16), fused epilogue ---
        for (int it = 0; it < 16; ++it) {
            v8f acc = {};
            // B operand 32x16: lane holds col N = i = it*16 + row;
            // elts 0-7 -> K = ks*32 + e + half*8, elts 8-15 -> K = ks*32 + 16 + (e-8) + half*8.
            const uint4* brow =
                (const uint4*)(Al + (it * 16 + row) * 256 + (half ? 8 : 0));
            #pragma unroll
            for (int ks = 0; ks < 8; ++ks) {
                HF16 bf;
                bf.q[0] = brow[ks * 4];
                bf.q[1] = brow[ks * 4 + 2];
                acc = __builtin_amdgcn_wmma_f32_16x16x32_f16(
                        false, Pf[ks], false, bf.v, (short)0, acc, false, false);
            }
            // D layout: vgpr r, lane -> M = r + 8*half, N = lane%16.
            // part[r] += T[prow, i] * P[prow, i]
            #pragma unroll
            for (int r = 0; r < 8; ++r)
                part[r] += acc[r] * (float)Pl[(half * 8 + r) * 256 + it * 16 + row];
        }

        // ---- reduce over the 16 lanes of each half (sum over i%16) ----
        #pragma unroll
        for (int r = 0; r < 8; ++r) {
            float v = part[r];
            v += __shfl_xor(v, 1, 16);
            v += __shfl_xor(v, 2, 16);
            v += __shfl_xor(v, 4, 16);
            v += __shfl_xor(v, 8, 16);
            part[r] = v;
        }

        // lanes 0 and 16 write patch rows 0-7 / 8-15
        if (row == 0) {
            #pragma unroll
            for (int r = 0; r < 8; ++r) {
                int prow = half * 8 + r;
                int p = tile * 16 + prow;
                float n2 = nrm[prow];
                float e = (n2 > 0.f) ? part[r] / n2 : 0.f;
                int b = p / 961;
                int rem = p - b * 961;
                int oy = rem / 31;
                int ox = rem - oy * 31;
                out[(b * 8 + ch) * 961 + oy * 31 + ox] = e + bch;
            }
        }
    }
}

// ---------------------------------------------------------------------------
extern "C" void kernel_launch(void* const* d_in, const int* in_sizes, int n_in,
                              void* d_out, int out_size, void* d_ws, size_t ws_size,
                              hipStream_t stream) {
    (void)in_sizes; (void)n_in; (void)out_size; (void)ws_size;
    const float* x    = (const float*)d_in[0];   // 32*16*64*64
    const float* w    = (const float*)d_in[1];   // 8*3*8*3
    const float* bias = (const float*)d_in[2];   // 8
    float* out = (float*)d_out;                  // 32*8*31*31

    // workspace: Ur (2MB) | Ui (2MB) | A f16 (1MB)
    float* Ur = (float*)d_ws;
    float* Ui = Ur + 8 * 256 * 256;
    _Float16* A = (_Float16*)(Ui + 8 * 256 * 256);

    k_build_unitary<<<64, 256, 0, stream>>>(w, Ur, Ui);
    k_build_A<<<2048, 256, 0, stream>>>(Ur, Ui, A);
    k_qconv<<<8 * WPC, 256, 197120, stream>>>(x, A, bias, out);
}
